// LSTMEncoder_80453327389129
// MI455X (gfx1250) — compile-verified, matrix operand order checked by reading, not run on hardware
//
#include <hip/hip_runtime.h>
#include <hip/hip_bf16.h>
#include <math.h>

typedef __bf16 bf16_t;
typedef __attribute__((ext_vector_type(16))) __bf16 v16bf;
typedef __attribute__((ext_vector_type(8)))  float  v8f;
typedef unsigned int v4u __attribute__((ext_vector_type(4)));
typedef int          v8i_t __attribute__((ext_vector_type(8)));
typedef int          v4i_t __attribute__((ext_vector_type(4)));

#define HDIM   1024
#define BATCH  16
#define SEQ    2048
#define GH     4096            // 4*HDIM
#define MROWS  (BATCH * SEQ)   // 32768
#define NWG    64              // workgroups in persistent scan kernel
#define NC     (HDIM / NWG)    // 16 hidden columns per workgroup

#if __has_builtin(__builtin_amdgcn_tensor_load_to_lds) && __has_builtin(__builtin_amdgcn_s_wait_tensorcnt)
#define LSTM_TDM 1
#endif

// ---------------------------------------------------------------------------
// Fragment loader: 16-bit A(16x32)/B(32x16) WMMA fragment for one lane.
// Per lane the 16 bf16 elements are two contiguous 8-element (16B) chunks:
// [k0 + 8*half, +8) and [k0 + 16 + 8*half, +8).   (ISA 7.12.2)
// ---------------------------------------------------------------------------
__device__ __forceinline__ v16bf lstm_load_frag(const bf16_t* row, int k0, int half) {
    union { v16bf v; uint4 q[2]; } u;
    u.q[0] = *reinterpret_cast<const uint4*>(row + k0 + half * 8);
    u.q[1] = *reinterpret_cast<const uint4*>(row + k0 + 16 + half * 8);
    return u.v;
}

__device__ __forceinline__ v8f lstm_wmma(v16bf a, v16bf b, v8f c) {
    return __builtin_amdgcn_wmma_f32_16x16x32_bf16(false, a, false, b,
                                                   (short)0, c, false, false);
}

__device__ __forceinline__ float lstm_sigmoid(float x) {
    return 1.0f / (1.0f + __expf(-x));
}

// ---------------------------------------------------------------------------
// Init: zero h double-buffer, c state, and the two per-layer barrier counters.
// ---------------------------------------------------------------------------
__global__ void lstm_init(bf16_t* hbuf, float* cst, int* ctr) {
    int i = blockIdx.x * blockDim.x + threadIdx.x;
    if (i < 2 * BATCH * HDIM) hbuf[i] = (bf16_t)0.0f;
    if (i < BATCH * HDIM)     cst[i]  = 0.0f;
    if (i < 2)                ctr[i]  = 0;
}

// f32 -> bf16 elementwise
__global__ void lstm_cvt_bf16(bf16_t* dst, const float* src, int n) {
    int i = blockIdx.x * blockDim.x + threadIdx.x;
    if (i < n) dst[i] = (bf16_t)src[i];
}

// Pack W4/U4 = concat(f,i,o,c) rows for one layer into bf16 [4096][1024].
__global__ void lstm_pack_w(bf16_t* dst, const float* wf, const float* wi,
                            const float* wo, const float* wc) {
    int i = blockIdx.x * blockDim.x + threadIdx.x;   // < 4*1024*1024
    int gate = i >> 20;
    int idx  = i & ((1 << 20) - 1);
    const float* s = (gate == 0) ? wf : (gate == 1) ? wi : (gate == 2) ? wo : wc;
    dst[i] = (bf16_t)s[idx];
}

// Pack b4 = concat(bf,bi,bo,bc) for one layer, f32 [4096].
__global__ void lstm_pack_b(float* dst, const float* bf_, const float* bi_,
                            const float* bo_, const float* bc_) {
    int i = blockIdx.x * blockDim.x + threadIdx.x;   // < 4096
    if (i >= GH) return;
    int gate = i >> 10;
    int idx  = i & (HDIM - 1);
    const float* s = (gate == 0) ? bf_ : (gate == 1) ? bi_ : (gate == 2) ? bo_ : bc_;
    dst[i] = s[idx];
}

// ---------------------------------------------------------------------------
// Input-projection GEMM:  xg[t][b][g] = sum_h x[b][t][h] * W4[g][h] + b4[g]
// Block = 128 threads (4 waves); block tile 16(M) x 256(N); each wave computes
// a 16x64 strip (4 accumulators, A fragment reused 4x).  The k-loop is a
// manual unroll-by-2 ping-pong: slot A lives across iterations, slot B is
// loop-local, so the double-buffering costs no register copies.
// ---------------------------------------------------------------------------
__global__ void lstm_gemm_xg(const bf16_t* __restrict__ xbf,
                             const bf16_t* __restrict__ w4,
                             const float*  __restrict__ b4,
                             float* __restrict__ xg) {
    int wave = threadIdx.x >> 5;
    int lane = threadIdx.x & 31;
    int half = lane >> 4;
    int m0 = blockIdx.x * 16;
    int nbase = blockIdx.y * 256 + wave * 64;

    const bf16_t* arow  = xbf + (size_t)(m0 + (lane & 15)) * HDIM;
    const bf16_t* brow0 = w4  + (size_t)(nbase + (lane & 15)) * HDIM;
    const bf16_t* brow1 = brow0 + 16 * HDIM;
    const bf16_t* brow2 = brow0 + 32 * HDIM;
    const bf16_t* brow3 = brow0 + 48 * HDIM;

    v8f acc0 = {}, acc1 = {}, acc2 = {}, acc3 = {};

    // Slot A (persists across iterations)
    v16bf aA  = lstm_load_frag(arow,  0, half);
    v16bf bA0 = lstm_load_frag(brow0, 0, half);
    v16bf bA1 = lstm_load_frag(brow1, 0, half);
    v16bf bA2 = lstm_load_frag(brow2, 0, half);
    v16bf bA3 = lstm_load_frag(brow3, 0, half);

    for (int k0 = 0; k0 < HDIM; k0 += 64) {
        // Slot B loads for k0+32 (always in range: HDIM % 64 == 0)
        v16bf aB  = lstm_load_frag(arow,  k0 + 32, half);
        v16bf bB0 = lstm_load_frag(brow0, k0 + 32, half);
        v16bf bB1 = lstm_load_frag(brow1, k0 + 32, half);
        v16bf bB2 = lstm_load_frag(brow2, k0 + 32, half);
        v16bf bB3 = lstm_load_frag(brow3, k0 + 32, half);

        acc0 = lstm_wmma(aA, bA0, acc0);
        acc1 = lstm_wmma(aA, bA1, acc1);
        acc2 = lstm_wmma(aA, bA2, acc2);
        acc3 = lstm_wmma(aA, bA3, acc3);

        if (k0 + 64 < HDIM) {   // refill slot A for k0+64
            aA  = lstm_load_frag(arow,  k0 + 64, half);
            bA0 = lstm_load_frag(brow0, k0 + 64, half);
            bA1 = lstm_load_frag(brow1, k0 + 64, half);
            bA2 = lstm_load_frag(brow2, k0 + 64, half);
            bA3 = lstm_load_frag(brow3, k0 + 64, half);
        }

        acc0 = lstm_wmma(aB, bB0, acc0);
        acc1 = lstm_wmma(aB, bB1, acc1);
        acc2 = lstm_wmma(aB, bB2, acc2);
        acc3 = lstm_wmma(aB, bB3, acc3);
    }

    v8f accs[4] = {acc0, acc1, acc2, acc3};
#pragma unroll
    for (int nt = 0; nt < 4; ++nt) {
        int col = nbase + nt * 16 + (lane & 15);
        float bias = b4[col];
#pragma unroll
        for (int v = 0; v < 8; ++v) {
            int mi = m0 + v + 8 * half;          // row = b*SEQ + t
            int b_ = mi >> 11;                   // SEQ = 2048
            int t  = mi & (SEQ - 1);
            xg[((size_t)t * BATCH + b_) * GH + col] = accs[nt][v] + bias;
        }
    }
}

// ---------------------------------------------------------------------------
// Persistent recurrent scan.  64 workgroups, each owns NC=16 hidden columns.
// U slice is DMA'd to LDS once (TDM when available), then each wave hoists its
// 16 timestep-invariant B fragments into registers -> the 2048-step loop does
// no LDS traffic at all: per step = 32 global b128 loads of h + 16 WMMAs.
// 8 waves: wave = gate(0..3) x khalf(0..1); K split 2x512, reduced via LDS;
// two interleaved accumulators halve the WMMA RAW chain.
// ---------------------------------------------------------------------------
__global__ void lstm_rec(const float*  __restrict__ xg,
                         const bf16_t* __restrict__ u4,
                         bf16_t* hbuf,            // [2][BATCH][HDIM] bf16
                         float*  cstate,          // [BATCH][HDIM] f32
                         float*  __restrict__ out,// [BATCH][SEQ][HDIM] f32
                         int*    ctr) {
    __shared__ __align__(16) bf16_t uld[4][NC][HDIM];   // 128 KB
    __shared__ float gls[4][BATCH][NC];                 // 4 KB
    __shared__ float cls[BATCH][NC];                    // 1 KB

    const int tid  = threadIdx.x;            // 256 threads = 8 waves
    const int wg   = blockIdx.x;
    const int n0   = wg * NC;
    const int wave = tid >> 5;
    const int gate = wave & 3;
    const int khalf = wave >> 2;
    const int lane = tid & 31;
    const int nl   = lane & 15;
    const int half = lane >> 4;

#ifdef LSTM_TDM
    // Tensor Data Mover: 4 x 2D tile loads (16 rows x 1024 bf16, row stride
    // 1024 elems) from u4 rows [g*1024+n0, +16) into uld[g].  One wave issues
    // the DMA (EXEC is ignored by TENSOR_LOAD_TO_LDS), then waits TENSORcnt.
    if (wave == 0) {
        unsigned int lds_base = (unsigned int)(uintptr_t)(&uld[0][0][0]);
#pragma unroll
        for (int g = 0; g < 4; ++g) {
            unsigned long long ga =
                (unsigned long long)(uintptr_t)(u4 + ((size_t)(g * HDIM + n0)) * HDIM);
            v4u g0;
            g0.x = 1u;                                             // count=1 (valid), user mode
            g0.y = lds_base + (unsigned int)(g * NC * HDIM * 2);   // lds_addr (bytes)
            g0.z = (unsigned int)ga;                               // global_addr[31:0]
            g0.w = (unsigned int)((ga >> 32) & 0x1FFFFFFu) | (2u << 30); // addr[56:32] | type=2
            v8i_t g1;
            g1[0] = (1 << 16);            // workgroup_mask=0, data_size=1 (2 bytes)
            g1[1] = (int)(1024u << 16);   // tensor_dim0[15:0]=1024 -> bits[63:48]
            g1[2] = (int)(16u << 16);     // tensor_dim0 hi=0 | tensor_dim1=16
            g1[3] = (int)(1024u << 16);   // tensor_dim1 hi=0 | tile_dim0=1024
            g1[4] = 16;                   // tile_dim1=16, tile_dim2=0
            g1[5] = 1024;                 // tensor_dim0_stride = 1024 elems
            g1[6] = 0;
            g1[7] = 0;
            v4i_t z4 = {0, 0, 0, 0};
#if defined(__clang_major__) && (__clang_major__ >= 23)
            v8i_t z8 = {0, 0, 0, 0, 0, 0, 0, 0};
            __builtin_amdgcn_tensor_load_to_lds(g0, g1, z4, z4, z8, 0);
#else
            __builtin_amdgcn_tensor_load_to_lds(g0, g1, z4, z4, 0);
#endif
        }
        __builtin_amdgcn_s_wait_tensorcnt(0);
    }
#else
    // Fallback: cooperative vector copy of the U slice.
    for (int i = tid; i < 4 * NC * HDIM / 8; i += 256) {
        int e = i * 8;
        int g = e >> 14;
        int rem = e & 16383;
        int r = rem >> 10;
        int k = rem & (HDIM - 1);
        *reinterpret_cast<uint4*>(&uld[g][r][k]) =
            *reinterpret_cast<const uint4*>(&u4[((size_t)(g * HDIM + n0 + r)) * HDIM + k]);
    }
#endif
    // Load carried c state
    {
        int b_ = tid >> 4, n = tid & 15;
        cls[b_][n] = cstate[b_ * HDIM + n0 + n];
    }
    __syncthreads();

    // Hoist this wave's 16 timestep-invariant B fragments LDS -> registers.
    const bf16_t* brow = &uld[gate][nl][0];
    const int kbeg = khalf * 512;
    v16bf bfr[16];
#pragma unroll
    for (int i = 0; i < 16; ++i)
        bfr[i] = lstm_load_frag(brow, kbeg + i * 32, half);

    for (int t = 0; t < SEQ; ++t) {
        const bf16_t* hread  = hbuf + (size_t)(t & 1) * BATCH * HDIM;
        bf16_t*       hwrite = hbuf + (size_t)((t + 1) & 1) * BATCH * HDIM;
        const bf16_t* arow = hread + (size_t)(lane & 15) * HDIM;   // row = batch

        v8f acc = {}, accB = {};
        v16bf a0 = lstm_load_frag(arow, kbeg, half);
        v16bf a1 = lstm_load_frag(arow, kbeg + 32, half);
#pragma unroll
        for (int i = 0; i < 16; i += 2) {
            v16bf na0 = a0, na1 = a1;
            if (i + 2 < 16) {
                na0 = lstm_load_frag(arow, kbeg + (i + 2) * 32, half);
                na1 = lstm_load_frag(arow, kbeg + (i + 3) * 32, half);
            }
            acc  = lstm_wmma(a0, bfr[i],     acc);
            accB = lstm_wmma(a1, bfr[i + 1], accB);
            a0 = na0; a1 = na1;
        }
#pragma unroll
        for (int v = 0; v < 8; ++v) acc[v] += accB[v];

        if (khalf == 1) {
#pragma unroll
            for (int v = 0; v < 8; ++v) gls[gate][v + 8 * half][nl] = acc[v];
        }
        __syncthreads();
        if (khalf == 0) {
            if (t + 1 < SEQ) {  // prefetch next step's xg slice into L2/L1
                __builtin_prefetch(&xg[((size_t)(t + 1) * BATCH) * GH + gate * HDIM + n0 + nl], 0, 1);
            }
#pragma unroll
            for (int v = 0; v < 8; ++v) {
                int b_ = v + 8 * half;
                float g = acc[v] + gls[gate][b_][nl]
                        + xg[((size_t)t * BATCH + b_) * GH + gate * HDIM + n0 + nl];
                g = (gate == 3) ? tanhf(g) : lstm_sigmoid(g);
                gls[gate][b_][nl] = g;
            }
        }
        __syncthreads();

        // Elementwise cell update: one (b, n) pair per thread.
        {
            int b_ = tid >> 4, n = tid & 15;
            float f  = gls[0][b_][n];
            float ii = gls[1][b_][n];
            float o  = gls[2][b_][n];
            float cc = gls[3][b_][n];
            float c  = f * cls[b_][n] + ii * cc;
            float h  = o * tanhf(c);
            cls[b_][n] = c;
            hwrite[b_ * HDIM + n0 + n] = (bf16_t)h;
            out[((size_t)b_ * SEQ + t) * HDIM + n0 + n] = h;
        }

        // Grid barrier: monotonic arrival counter (NWG blocks co-resident).
        __syncthreads();
        if (tid == 0) {
            __threadfence();
            __hip_atomic_fetch_add(ctr, 1, __ATOMIC_RELEASE, __HIP_MEMORY_SCOPE_AGENT);
            int target = NWG * (t + 1);
            while (__hip_atomic_load(ctr, __ATOMIC_ACQUIRE, __HIP_MEMORY_SCOPE_AGENT) < target) {
                __builtin_amdgcn_s_sleep(2);
            }
        }
        __syncthreads();
    }

    // Spill final c so the next layer inherits (h_T, c_T).
    {
        int b_ = tid >> 4, n = tid & 15;
        cstate[b_ * HDIM + n0 + n] = cls[b_][n];
    }
}

// ---------------------------------------------------------------------------
// Host-side orchestration.
// Inputs: 0:x 1:Wf 2:Uf 3:Wi 4:Ui 5:Wo 6:Uo 7:Wc 8:Uc 9:bf 10:bi 11:bo 12:bc
// ---------------------------------------------------------------------------
extern "C" void kernel_launch(void* const* d_in, const int* in_sizes, int n_in,
                              void* d_out, int out_size, void* d_ws, size_t ws_size,
                              hipStream_t stream) {
    (void)in_sizes; (void)n_in; (void)out_size; (void)ws_size;

    const float* x  = (const float*)d_in[0];
    const float* Wf = (const float*)d_in[1];
    const float* Uf = (const float*)d_in[2];
    const float* Wi = (const float*)d_in[3];
    const float* Ui = (const float*)d_in[4];
    const float* Wo = (const float*)d_in[5];
    const float* Uo = (const float*)d_in[6];
    const float* Wc = (const float*)d_in[7];
    const float* Uc = (const float*)d_in[8];
    const float* bF = (const float*)d_in[9];
    const float* bI = (const float*)d_in[10];
    const float* bO = (const float*)d_in[11];
    const float* bC = (const float*)d_in[12];

    char* ws = (char*)d_ws;
    size_t off = 0;
    float*  xg   = (float*)(ws + off);  off += (size_t)SEQ * BATCH * GH * 4;     // 512 MB
    float*  x1   = (float*)(ws + off);  off += (size_t)BATCH * SEQ * HDIM * 4;   // 128 MB
    bf16_t* xbf  = (bf16_t*)(ws + off); off += (size_t)BATCH * SEQ * HDIM * 2;   // 64 MB
    bf16_t* w4   = (bf16_t*)(ws + off); off += (size_t)GH * HDIM * 2;            // 8 MB
    bf16_t* u4   = (bf16_t*)(ws + off); off += (size_t)GH * HDIM * 2;            // 8 MB
    float*  b4   = (float*)(ws + off);  off += (size_t)GH * 4;
    bf16_t* hbuf = (bf16_t*)(ws + off); off += (size_t)2 * BATCH * HDIM * 2;
    float*  cst  = (float*)(ws + off);  off += (size_t)BATCH * HDIM * 4;
    int*    ctr  = (int*)(ws + off);    off += 256;

    const int nX = BATCH * SEQ * HDIM;          // 33554432
    const int nW = GH * HDIM;                   // 4194304

    lstm_init<<<(2 * BATCH * HDIM + 255) / 256, 256, 0, stream>>>(hbuf, cst, ctr);

    for (int layer = 0; layer < 2; ++layer) {
        size_t woff = (size_t)layer * HDIM * HDIM;
        size_t boff = (size_t)layer * HDIM;
        const float* xin = (layer == 0) ? x : x1;
        float* lout = (layer == 0) ? x1 : (float*)d_out;

        lstm_cvt_bf16<<<nX / 256, 256, 0, stream>>>(xbf, xin, nX);
        lstm_pack_w<<<nW / 256, 256, 0, stream>>>(w4, Wf + woff, Wi + woff, Wo + woff, Wc + woff);
        lstm_pack_w<<<nW / 256, 256, 0, stream>>>(u4, Uf + woff, Ui + woff, Uo + woff, Uc + woff);
        lstm_pack_b<<<GH / 256, 256, 0, stream>>>(b4, bF + boff, bI + boff, bO + boff, bC + boff);

        dim3 ggrid(MROWS / 16, GH / 256, 1);
        lstm_gemm_xg<<<ggrid, 128, 0, stream>>>(xbf, w4, b4, xg);

        lstm_rec<<<NWG, 256, 0, stream>>>(xg, u4, hbuf, cst, lout, ctr + layer);
    }
}
